// CalculateAttention_7722351198508
// MI455X (gfx1250) — matrix operand order repaired and checked by reading, not compiled, not run on hardware
//
#include <hip/hip_runtime.h>

// ---------------------------------------------------------------------------
// Fused dual-modal flash attention for MI455X (gfx1250, wave32, WMMA).
//   scores = 0.5*(Qx Kx^T + Qy Ky^T) / sqrt(64);  P = softmax(scores)
//   out1 = P Vx ; out2 = P Vy
// bf16 WMMA (v_wmma_f32_16x16x32_bf16) with fp32 accumulate, online softmax.
// v3: B/V fragments batch-loaded ahead of each WMMA chain (ds loads issue in
//     bulk, WMMAs run back-to-back instead of waiting dscnt each), P-bounce
//     stores overlapped with accumulator rescale. Double-buffered K/V tiles,
//     Q pre-scaled, row-sum reduction deferred to epilogue.
// ---------------------------------------------------------------------------

#define B_DIM 2
#define H_DIM 8
#define S_DIM 2048
#define D_DIM 64

#define BLOCK_THREADS   256
#define WAVES_PER_BLOCK 8
#define ROWS_PER_WAVE   16
#define ROWS_PER_BLOCK  128
#define TK              32     // keys per tile
#define NITER           (S_DIM / TK)

typedef __attribute__((ext_vector_type(16))) __bf16 v16bf;
typedef __attribute__((ext_vector_type(8)))  __bf16 v8bf;
typedef __attribute__((ext_vector_type(8)))  float  v8f;

static __device__ __forceinline__ v16bf cat8(v8bf lo, v8bf hi) {
  return __builtin_shufflevector(lo, hi, 0,1,2,3,4,5,6,7,8,9,10,11,12,13,14,15);
}

// A-operand fragment (16x32 bf16): lane holds row m = lane&15,
// k in [k0,k0+8) and [k0+16,k0+24), k0 = (lane<16 ? 0 : 8).
static __device__ __forceinline__ v16bf lds_frag_a(const __bf16* p) {
  v8bf lo = *(const v8bf*)(p);
  v8bf hi = *(const v8bf*)(p + 16);
  return cat8(lo, hi);
}

// B-operand fragment (32x16 bf16): lane holds column n = lane&15, 16 contiguous k values from p.
static __device__ __forceinline__ v16bf lds_frag_b(const __bf16* p) {
  v8bf lo = *(const v8bf*)(p);
  v8bf hi = *(const v8bf*)(p + 8);
  return cat8(lo, hi);
}

// A fragment from global fp32 (row-major), scaled by s before bf16 convert.
static __device__ __forceinline__ v16bf gbl_frag_a_f32s(const float* p, float s) {
  float4 c0 = ((const float4*)p)[0];
  float4 c1 = ((const float4*)p)[1];
  float4 c2 = ((const float4*)p)[4];   // p + 16 floats
  float4 c3 = ((const float4*)p)[5];
  v16bf a;
  a[0]=(__bf16)(c0.x*s);  a[1]=(__bf16)(c0.y*s);  a[2]=(__bf16)(c0.z*s);  a[3]=(__bf16)(c0.w*s);
  a[4]=(__bf16)(c1.x*s);  a[5]=(__bf16)(c1.y*s);  a[6]=(__bf16)(c1.z*s);  a[7]=(__bf16)(c1.w*s);
  a[8]=(__bf16)(c2.x*s);  a[9]=(__bf16)(c2.y*s);  a[10]=(__bf16)(c2.z*s); a[11]=(__bf16)(c2.w*s);
  a[12]=(__bf16)(c3.x*s); a[13]=(__bf16)(c3.y*s); a[14]=(__bf16)(c3.z*s); a[15]=(__bf16)(c3.w*s);
  return a;
}

// Cooperative staging: K row-major [key][d], V transposed [d][key], fp32->bf16.
static __device__ __forceinline__ void stage_tiles(
    const float* __restrict__ Kx, const float* __restrict__ Ky,
    const float* __restrict__ Vx, const float* __restrict__ Vy,
    int kt, int srow, int scol,
    __bf16* dKTx, __bf16* dKTy, __bf16* dVx, __bf16* dVy)
{
  const size_t go = (size_t)(kt + srow) * D_DIM + scol;
  float4 kx0 = ((const float4*)(Kx + go))[0];
  float4 kx1 = ((const float4*)(Kx + go))[1];
  float4 ky0 = ((const float4*)(Ky + go))[0];
  float4 ky1 = ((const float4*)(Ky + go))[1];
  float4 vx0 = ((const float4*)(Vx + go))[0];
  float4 vx1 = ((const float4*)(Vx + go))[1];
  float4 vy0 = ((const float4*)(Vy + go))[0];
  float4 vy1 = ((const float4*)(Vy + go))[1];
  float fx[8] = {kx0.x,kx0.y,kx0.z,kx0.w,kx1.x,kx1.y,kx1.z,kx1.w};
  float fy[8] = {ky0.x,ky0.y,ky0.z,ky0.w,ky1.x,ky1.y,ky1.z,ky1.w};
  #pragma unroll
  for (int i = 0; i < 8; ++i) {
    dKTx[srow * D_DIM + scol + i] = (__bf16)fx[i];
    dKTy[srow * D_DIM + scol + i] = (__bf16)fy[i];
  }
  v8bf pvx, pvy;
  pvx[0]=(__bf16)vx0.x; pvx[1]=(__bf16)vx0.y; pvx[2]=(__bf16)vx0.z; pvx[3]=(__bf16)vx0.w;
  pvx[4]=(__bf16)vx1.x; pvx[5]=(__bf16)vx1.y; pvx[6]=(__bf16)vx1.z; pvx[7]=(__bf16)vx1.w;
  pvy[0]=(__bf16)vy0.x; pvy[1]=(__bf16)vy0.y; pvy[2]=(__bf16)vy0.z; pvy[3]=(__bf16)vy0.w;
  pvy[4]=(__bf16)vy1.x; pvy[5]=(__bf16)vy1.y; pvy[6]=(__bf16)vy1.z; pvy[7]=(__bf16)vy1.w;
  #pragma unroll
  for (int i = 0; i < 8; ++i) {
    dVx[(scol + i) * TK + srow] = pvx[i];
    dVy[(scol + i) * TK + srow] = pvy[i];
  }
}

__global__ __launch_bounds__(BLOCK_THREADS)
void dual_flash_attn_kernel(const float* __restrict__ gQx, const float* __restrict__ gKx,
                            const float* __restrict__ gVx, const float* __restrict__ gQy,
                            const float* __restrict__ gKy, const float* __restrict__ gVy,
                            float* __restrict__ gO1, float* __restrict__ gO2)
{
  __shared__ __bf16 sKTx[2][D_DIM * TK];   // K^T tiles, double buffered
  __shared__ __bf16 sKTy[2][D_DIM * TK];
  __shared__ __bf16 sVx [2][TK * D_DIM];   // V tiles transposed [d][key], double buffered
  __shared__ __bf16 sVy [2][TK * D_DIM];
  __shared__ __bf16 sP  [WAVES_PER_BLOCK * ROWS_PER_WAVE * TK]; // per-wave P bounce

  const int tid   = threadIdx.x;
  const int lane  = tid & 31;
  const int wave  = tid >> 5;
  const int col   = lane & 15;
  const int rhalf = (lane < 16) ? 0 : 8;  // C/D row offset for this half-wave
  const int klo   = rhalf;                // A-operand k base for this half-wave

  const int bh      = blockIdx.y;
  const size_t base = (size_t)bh * S_DIM * D_DIM;
  const float* Qx = gQx + base;  const float* Kx = gKx + base;  const float* Vx = gVx + base;
  const float* Qy = gQy + base;  const float* Ky = gKy + base;  const float* Vy = gVy + base;

  const int qbase = blockIdx.x * ROWS_PER_BLOCK + wave * ROWS_PER_WAVE;

  // Q fragments: load once, pre-scaled by 0.5*rsqrt(D), fp32 -> bf16.
  constexpr float SCALE = 0.0625f;
  const float* qxr = Qx + (size_t)(qbase + col) * D_DIM + klo;
  const float* qyr = Qy + (size_t)(qbase + col) * D_DIM + klo;
  v16bf aqx[2], aqy[2];
  aqx[0] = gbl_frag_a_f32s(qxr, SCALE);      aqx[1] = gbl_frag_a_f32s(qxr + 32, SCALE);
  aqy[0] = gbl_frag_a_f32s(qyr, SCALE);      aqy[1] = gbl_frag_a_f32s(qyr + 32, SCALE);

  v8f acc1[4] = {};   // P @ Vx, d-chunks 0..3
  v8f acc2[4] = {};   // P @ Vy
  float mrow[8], lrow[8];   // lrow = per-lane PARTIAL sums (reduced at the end)
  #pragma unroll
  for (int r = 0; r < 8; ++r) { mrow[r] = -1.0e30f; lrow[r] = 0.0f; }

  __bf16* pw = &sP[wave * ROWS_PER_WAVE * TK];

  const int srow = tid >> 3;         // cooperative staging: row 0..31
  const int scol = (tid & 7) << 3;   // col 0,8,..,56 (8 floats each)

  // Prologue: stage tile 0 into buffer 0.
  stage_tiles(Kx, Ky, Vx, Vy, 0, srow, scol, sKTx[0], sKTy[0], sVx[0], sVy[0]);
  __syncthreads();

  for (int it = 0; it < NITER; ++it) {
    const int kt  = it * TK;
    const int cur = it & 1;

    // Stage next tile into the alternate buffer (overlaps with WMMA below).
    if (it + 1 < NITER)
      stage_tiles(Kx, Ky, Vx, Vy, kt + TK, srow, scol,
                  sKTx[cur ^ 1], sKTy[cur ^ 1], sVx[cur ^ 1], sVy[cur ^ 1]);
    if (it + 2 < NITER) {   // prefetch tile i+2 (global_prefetch_b8)
      const size_t gn = (size_t)(kt + 2 * TK + srow) * D_DIM + scol;
      __builtin_prefetch(Kx + gn, 0, 0);
      __builtin_prefetch(Ky + gn, 0, 0);
      __builtin_prefetch(Vx + gn, 0, 0);
      __builtin_prefetch(Vy + gn, 0, 0);
    }

    const __bf16* tKTx = sKTx[cur];
    const __bf16* tKTy = sKTy[cur];
    const __bf16* tVx  = sVx[cur];
    const __bf16* tVy  = sVy[cur];

    // ---- batch-load all 8 score B fragments, then run the WMMA chain ----
    const int d0 = col * D_DIM + klo;          // K row key=col, d-chunk 0
    const int d1 = col * D_DIM + 32 + klo;     // K row key=col, d-chunk 1
    v16bf bx00 = lds_frag_a(&tKTx[d0]);
    v16bf bx01 = lds_frag_a(&tKTx[d0 + 16 * D_DIM]);
    v16bf by00 = lds_frag_a(&tKTy[d0]);
    v16bf by01 = lds_frag_a(&tKTy[d0 + 16 * D_DIM]);
    v16bf bx10 = lds_frag_a(&tKTx[d1]);
    v16bf bx11 = lds_frag_a(&tKTx[d1 + 16 * D_DIM]);
    v16bf by10 = lds_frag_a(&tKTy[d1]);
    v16bf by11 = lds_frag_a(&tKTy[d1 + 16 * D_DIM]);

    v8f sc0 = {}, sc1 = {};   // keys [kt,kt+16) / [kt+16,kt+32)
    sc0 = __builtin_amdgcn_wmma_f32_16x16x32_bf16(false, aqx[0], false, bx00, (short)0, sc0, false, false);
    sc1 = __builtin_amdgcn_wmma_f32_16x16x32_bf16(false, aqx[0], false, bx01, (short)0, sc1, false, false);
    sc0 = __builtin_amdgcn_wmma_f32_16x16x32_bf16(false, aqy[0], false, by00, (short)0, sc0, false, false);
    sc1 = __builtin_amdgcn_wmma_f32_16x16x32_bf16(false, aqy[0], false, by01, (short)0, sc1, false, false);
    sc0 = __builtin_amdgcn_wmma_f32_16x16x32_bf16(false, aqx[1], false, bx10, (short)0, sc0, false, false);
    sc1 = __builtin_amdgcn_wmma_f32_16x16x32_bf16(false, aqx[1], false, bx11, (short)0, sc1, false, false);
    sc0 = __builtin_amdgcn_wmma_f32_16x16x32_bf16(false, aqy[1], false, by10, (short)0, sc0, false, false);
    sc1 = __builtin_amdgcn_wmma_f32_16x16x32_bf16(false, aqy[1], false, by11, (short)0, sc1, false, false);

    // ---- online softmax (rows live in 16-lane halves of C layout) ----
    constexpr float L2E = 1.4426950408889634f;
    float corr8[8];
    #pragma unroll
    for (int r = 0; r < 8; ++r) {
      float s0 = sc0[r];                 // Q was pre-scaled: these ARE the scores
      float s1 = sc1[r];
      float mx = fmaxf(s0, s1);
      mx = fmaxf(mx, __shfl_xor(mx, 1, 32));
      mx = fmaxf(mx, __shfl_xor(mx, 2, 32));
      mx = fmaxf(mx, __shfl_xor(mx, 4, 32));
      mx = fmaxf(mx, __shfl_xor(mx, 8, 32));
      const float mnew = fmaxf(mrow[r], mx);
      const float corr = exp2f((mrow[r] - mnew) * L2E);
      const float p0 = exp2f((s0 - mnew) * L2E);
      const float p1 = exp2f((s1 - mnew) * L2E);
      lrow[r] = lrow[r] * corr + (p0 + p1);   // per-lane partial; reduce at end
      mrow[r] = mnew;
      corr8[r] = corr;
      sc0[r] = p0;
      sc1[r] = p1;
    }

    // ---- P -> per-wave LDS first (store latency hides under rescale) ----
    #pragma unroll
    for (int r = 0; r < 8; ++r) {
      pw[(r + rhalf) * TK + col]      = (__bf16)sc0[r];
      pw[(r + rhalf) * TK + col + 16] = (__bf16)sc1[r];
    }

    // ---- rescale output accumulators while the P stores land ----
    #pragma unroll
    for (int c = 0; c < 4; ++c) {
      #pragma unroll
      for (int r = 0; r < 8; ++r) {
        acc1[c][r] = acc1[c][r] * corr8[r];
        acc2[c][r] = acc2[c][r] * corr8[r];
      }
    }

    // ---- batch-load P (A-layout) + all 8 V fragments, then PV chain ----
    v16bf ap = lds_frag_a(&pw[col * TK + klo]);   // in-order LDS within wave
    v16bf bvx0 = lds_frag_a(&tVx[( 0 + col) * TK + klo]);
    v16bf bvx1 = lds_frag_a(&tVx[(16 + col) * TK + klo]);
    v16bf bvx2 = lds_frag_a(&tVx[(32 + col) * TK + klo]);
    v16bf bvx3 = lds_frag_a(&tVx[(48 + col) * TK + klo]);
    v16bf bvy0 = lds_frag_a(&tVy[( 0 + col) * TK + klo]);
    v16bf bvy1 = lds_frag_a(&tVy[(16 + col) * TK + klo]);
    v16bf bvy2 = lds_frag_a(&tVy[(32 + col) * TK + klo]);
    v16bf bvy3 = lds_frag_a(&tVy[(48 + col) * TK + klo]);

    acc1[0] = __builtin_amdgcn_wmma_f32_16x16x32_bf16(false, ap, false, bvx0, (short)0, acc1[0], false, false);
    acc1[1] = __builtin_amdgcn_wmma_f32_16x16x32_bf16(false, ap, false, bvx1, (short)0, acc1[1], false, false);
    acc1[2] = __builtin_amdgcn_wmma_f32_16x16x32_bf16(false, ap, false, bvx2, (short)0, acc1[2], false, false);
    acc1[3] = __builtin_amdgcn_wmma_f32_16x16x32_bf16(false, ap, false, bvx3, (short)0, acc1[3], false, false);
    acc2[0] = __builtin_amdgcn_wmma_f32_16x16x32_bf16(false, ap, false, bvy0, (short)0, acc2[0], false, false);
    acc2[1] = __builtin_amdgcn_wmma_f32_16x16x32_bf16(false, ap, false, bvy1, (short)0, acc2[1], false, false);
    acc2[2] = __builtin_amdgcn_wmma_f32_16x16x32_bf16(false, ap, false, bvy2, (short)0, acc2[2], false, false);
    acc2[3] = __builtin_amdgcn_wmma_f32_16x16x32_bf16(false, ap, false, bvy3, (short)0, acc2[3], false, false);

    __syncthreads();   // next buffer staged; everyone done reading cur buffer
  }

  // ---- epilogue: reduce row sums across the 16-lane group, normalize ----
  #pragma unroll
  for (int r = 0; r < 8; ++r) {
    float ls = lrow[r];
    ls += __shfl_xor(ls, 1, 32);
    ls += __shfl_xor(ls, 2, 32);
    ls += __shfl_xor(ls, 4, 32);
    ls += __shfl_xor(ls, 8, 32);
    const float inv = 1.0f / ls;
    const size_t row = (size_t)(qbase + rhalf + r);
    float* o1 = gO1 + base + row * D_DIM;
    float* o2 = gO2 + base + row * D_DIM;
    #pragma unroll
    for (int c = 0; c < 4; ++c) {
      o1[16 * c + col] = acc1[c][r] * inv;
      o2[16 * c + col] = acc2[c][r] * inv;
    }
  }
}

extern "C" void kernel_launch(void* const* d_in, const int* in_sizes, int n_in,
                              void* d_out, int out_size, void* d_ws, size_t ws_size,
                              hipStream_t stream) {
  (void)in_sizes; (void)n_in; (void)out_size; (void)d_ws; (void)ws_size;
  const float* Qx = (const float*)d_in[0];
  const float* Kx = (const float*)d_in[1];
  const float* Vx = (const float*)d_in[2];
  const float* Qy = (const float*)d_in[3];
  const float* Ky = (const float*)d_in[4];
  const float* Vy = (const float*)d_in[5];
  float* O1 = (float*)d_out;
  float* O2 = O1 + (size_t)B_DIM * H_DIM * S_DIM * D_DIM;

  dim3 grid(S_DIM / ROWS_PER_BLOCK, B_DIM * H_DIM);
  dim3 block(BLOCK_THREADS);
  dual_flash_attn_kernel<<<grid, block, 0, stream>>>(Qx, Kx, Vx, Qy, Ky, Vy, O1, O2);
}
